// ConditionalRoutedAttention_51024211476836
// MI455X (gfx1250) — matrix operand order repaired and checked by reading, not compiled, not run on hardware
//
#include <hip/hip_runtime.h>
#include <hip/hip_bf16.h>

// ---------------------------------------------------------------------------
// Conditional Routed Attention for MI455X (gfx1250), wave32 + WMMA f16.
// ---------------------------------------------------------------------------

#define DIMC      1024
#define HEADS_C   8
#define DH_C      64
#define SEQ_C     8192
#define BATCH_C   2
#define WIN_C     64
#define NW_C      (SEQ_C / WIN_C)      // 128 windows
#define NQ_C      1024
#define HD_C      (HEADS_C * DH_C)     // 512
#define QKV_C     (3 * HD_C)           // 1536
#define KT_PAD    1056                 // 1 null + 1024 keys, padded to 66 tiles

typedef __attribute__((ext_vector_type(16))) _Float16 v16h;
typedef __attribute__((ext_vector_type(8)))  float    v8f;

// ------------------------- WMMA helpers ------------------------------------

__device__ __forceinline__ v8f wmma_f16(v16h a, v16h b, v8f c) {
  // D = A(16x32,f16) * B(32x16,f16) + C(16x16,f32)
  return __builtin_amdgcn_wmma_f32_16x16x32_f16(false, a, false, b, (short)0, c,
                                                false, false);
}

// A fragment: 16x32 f16, row-major source, lda elements per row.
__device__ __forceinline__ v16h frag_a_f16(const _Float16* A, int lda) {
  const int lane = threadIdx.x & 31;
  const int m  = lane & 15;
  const int kb = (lane >> 4) << 4;                    // 0 or 16
  v16h a;
#pragma unroll
  for (int v = 0; v < 8; ++v) {
    const int k = kb + 2 * v;
    a[2 * v]     = A[(size_t)m * lda + k];
    a[2 * v + 1] = A[(size_t)m * lda + k + 1];
  }
  return a;
}

// A fragment sourced from f32 LDS (probabilities), converted to f16.
__device__ __forceinline__ v16h frag_a_from_f32(const float* S, int lds) {
  const int lane = threadIdx.x & 31;
  const int m  = lane & 15;
  const int kb = (lane >> 4) << 4;
  v16h a;
#pragma unroll
  for (int v = 0; v < 8; ++v) {
    const int k = kb + 2 * v;
    a[2 * v]     = (_Float16)S[m * lds + k];
    a[2 * v + 1] = (_Float16)S[m * lds + k + 1];
  }
  return a;
}

// B fragment: 32x16 f16 from row-major KxN source.
__device__ __forceinline__ v16h frag_b_f16(const _Float16* B, int ldb) {
  const int lane = threadIdx.x & 31;
  const int n  = lane & 15;
  const int kb = (lane >> 4) << 4;                    // 0 or 16
  v16h b;
#pragma unroll
  for (int v = 0; v < 8; ++v) {
    b[2 * v]     = B[(size_t)(kb + 2 * v) * ldb + n];
    b[2 * v + 1] = B[(size_t)(kb + 2 * v + 1) * ldb + n];
  }
  return b;
}

// B fragment from transposed (row-major NxK) source.
__device__ __forceinline__ v16h frag_bt_f16(const _Float16* Bt, int ldb) {
  const int lane = threadIdx.x & 31;
  const int n  = lane & 15;
  const int kb = (lane >> 4) << 4;
  v16h b;
#pragma unroll
  for (int v = 0; v < 8; ++v) {
    b[2 * v]     = Bt[(size_t)n * ldb + kb + 2 * v];
    b[2 * v + 1] = Bt[(size_t)n * ldb + kb + 2 * v + 1];
  }
  return b;
}

// ------------------------- elementwise / prep kernels -----------------------

__global__ void f32_to_f16_kernel(const float* __restrict__ src,
                                  _Float16* __restrict__ dst, int n) {
  int i = blockIdx.x * blockDim.x + threadIdx.x;
  if (i < n) dst[i] = (_Float16)src[i];
}

__global__ void zero_f16_kernel(_Float16* __restrict__ dst, int n) {
  int i = blockIdx.x * blockDim.x + threadIdx.x;
  if (i < n) dst[i] = (_Float16)0.0f;
}

__global__ __launch_bounds__(256)
void ln_kernel(const float* __restrict__ x, const float* __restrict__ g,
               const float* __restrict__ be, _Float16* __restrict__ xn) {
  __shared__ float r1[256], r2[256];
  const size_t tok = blockIdx.x;
  const float* xp = x + tok * DIMC;
  float v[4], s1 = 0.f, s2 = 0.f;
#pragma unroll
  for (int j = 0; j < 4; ++j) {
    v[j] = xp[threadIdx.x + j * 256];
    s1 += v[j]; s2 += v[j] * v[j];
  }
  r1[threadIdx.x] = s1; r2[threadIdx.x] = s2;
  __syncthreads();
  for (int o = 128; o > 0; o >>= 1) {
    if (threadIdx.x < o) {
      r1[threadIdx.x] += r1[threadIdx.x + o];
      r2[threadIdx.x] += r2[threadIdx.x + o];
    }
    __syncthreads();
  }
  const float mu  = r1[0] * (1.0f / DIMC);
  const float var = r2[0] * (1.0f / DIMC) - mu * mu;
  const float rstd = rsqrtf(var + 1e-5f);
#pragma unroll
  for (int j = 0; j < 4; ++j) {
    const int i = threadIdx.x + j * 256;
    xn[tok * DIMC + i] = (_Float16)((v[j] - mu) * rstd * g[i] + be[i]);
  }
}

__global__ __launch_bounds__(256)
void route_score_kernel(const float* __restrict__ x, const float* __restrict__ tq,
                        const float* __restrict__ tkv, float* __restrict__ sq,
                        float* __restrict__ skv) {
  __shared__ float r1[256], r2[256];
  const size_t tok = blockIdx.x;
  const float* xp = x + tok * DIMC;
  float a1 = 0.f, a2 = 0.f;
#pragma unroll
  for (int j = 0; j < 4; ++j) {
    const int i = threadIdx.x + j * 256;
    const float xv = xp[i];
    a1 += xv * tq[i]; a2 += xv * tkv[i];
  }
  r1[threadIdx.x] = a1; r2[threadIdx.x] = a2;
  __syncthreads();
  for (int o = 128; o > 0; o >>= 1) {
    if (threadIdx.x < o) {
      r1[threadIdx.x] += r1[threadIdx.x + o];
      r2[threadIdx.x] += r2[threadIdx.x + o];
    }
    __syncthreads();
  }
  if (threadIdx.x == 0) { sq[tok] = r1[0]; skv[tok] = r2[0]; }
}

// Coordinate descent (forward): 20 iterations of per-batch logsumexp.
__global__ __launch_bounds__(1024)
void coor_descent_kernel(const float* __restrict__ s_in, float* __restrict__ sc_out) {
  __shared__ float red[1024];
  const int b = blockIdx.x, t = threadIdx.x;
  float s[8], bb[8];
#pragma unroll
  for (int j = 0; j < 8; ++j) {
    s[j] = s_in[(size_t)b * SEQ_C + t + j * 1024];
    bb[j] = -s[j];
  }
  float a = 0.f, cur = 4.0f;                 // EPS_INIT
  const float logk = logf(1152.0f);          // NQ * 9/8
  for (int it = 0; it < 20; ++it) {
    const float inv = 1.0f / cur;
    float lm = -3.4e38f;
#pragma unroll
    for (int j = 0; j < 8; ++j) lm = fmaxf(lm, (s[j] + bb[j]) * inv);
    red[t] = lm; __syncthreads();
    for (int o = 512; o > 0; o >>= 1) {
      if (t < o) red[t] = fmaxf(red[t], red[t + o]);
      __syncthreads();
    }
    const float M = red[0]; __syncthreads();
    float ls = 0.f;
#pragma unroll
    for (int j = 0; j < 8; ++j) ls += __expf((s[j] + bb[j]) * inv - M);
    red[t] = ls; __syncthreads();
    for (int o = 512; o > 0; o >>= 1) {
      if (t < o) red[t] += red[t + o];
      __syncthreads();
    }
    const float lse = logf(red[0]) + M; __syncthreads();
    a = cur * (logk - lse);
#pragma unroll
    for (int j = 0; j < 8; ++j) bb[j] = -fmaxf(s[j] + a, 0.f);
    cur = fmaxf(cur * 0.7f, 0.03f);
  }
  const float inv = 1.0f / cur;
#pragma unroll
  for (int j = 0; j < 8; ++j)
    sc_out[(size_t)b * SEQ_C + t + j * 1024] = __expf((s[j] + a + bb[j]) * inv);
}

// Per-batch bitonic top-k: sort (score desc, idx asc) via 64-bit keys in LDS.
__global__ __launch_bounds__(1024)
void topk_sort_kernel(const float* __restrict__ scores, int* __restrict__ sel_idx) {
  __shared__ unsigned long long key[SEQ_C];       // 64 KB
  const int b = blockIdx.x, t = threadIdx.x;
  for (int i = t; i < SEQ_C; i += 1024) {
    unsigned int bits = __float_as_uint(scores[(size_t)b * SEQ_C + i]);
    bits = (bits & 0x80000000u) ? ~bits : (bits | 0x80000000u);
    key[i] = ((unsigned long long)bits << 13) |
             (unsigned long long)(SEQ_C - 1 - i);
  }
  __syncthreads();
  for (int k = 2; k <= SEQ_C; k <<= 1) {
    for (int j = k >> 1; j > 0; j >>= 1) {
      for (int q = t; q < SEQ_C / 2; q += 1024) {
        const int i = ((q / j) * (2 * j)) + (q % j);
        const int p = i + j;
        const unsigned long long ai = key[i], ap = key[p];
        const bool up = ((i & k) == 0);
        const bool sw = up ? (ai < ap) : (ai > ap);   // descending sort
        if (sw) { key[i] = ap; key[p] = ai; }
      }
      __syncthreads();
    }
  }
  for (int i = t; i < NQ_C; i += 1024)
    sel_idx[b * NQ_C + i] = SEQ_C - 1 - (int)(key[i] & 8191ULL);
}

__global__ __launch_bounds__(256)
void gather_rmsnorm_kernel(const float* __restrict__ x, const int* __restrict__ sel,
                           const float* __restrict__ gamma, _Float16* __restrict__ out) {
  __shared__ float red[256];
  const int slot = blockIdx.x;                 // b*NQ + i
  const int b = slot >> 10;
  const int tok = sel[slot];
  const float* xp = x + ((size_t)b * SEQ_C + tok) * DIMC;
  float v[4], ss = 0.f;
#pragma unroll
  for (int j = 0; j < 4; ++j) {
    v[j] = xp[threadIdx.x + j * 256];
    ss += v[j] * v[j];
  }
  red[threadIdx.x] = ss; __syncthreads();
  for (int o = 128; o > 0; o >>= 1) {
    if (threadIdx.x < o) red[threadIdx.x] += red[threadIdx.x + o];
    __syncthreads();
  }
  const float nrm = sqrtf(red[0]);
  const float sc = 32.0f / fmaxf(nrm, 1e-12f);   // sqrt(1024) / max(||x||,eps)
#pragma unroll
  for (int j = 0; j < 4; ++j) {
    const int i = threadIdx.x + j * 256;
    out[(size_t)slot * DIMC + i] = (_Float16)(v[j] * sc * gamma[i]);
  }
}

// ------------------------- generic WMMA GEMM --------------------------------
// C(MxN,f32|f16) = A(MxK,f16 row-major) * B(KxN,f16; BT -> NxK row-major)

template <bool BT, bool O16>
__global__ __launch_bounds__(256)
void gemm_wmma_kernel(const _Float16* __restrict__ A, int lda,
                      const _Float16* __restrict__ B, int ldb,
                      void* __restrict__ C, int ldc, int Ktot) {
  const int wv = threadIdx.x >> 5, lane = threadIdx.x & 31;
  const int n0 = blockIdx.x * 16;
  const int m0 = (blockIdx.y * 8 + wv) * 16;
  const _Float16* Ap = A + (size_t)m0 * lda;
  v8f c = {};
  for (int k0 = 0; k0 < Ktot; k0 += 32) {
    __builtin_prefetch(Ap + k0 + 128, 0, 1);   // global_prefetch_b8
    v16h a = frag_a_f16(Ap + k0, lda);
    v16h bf;
    if constexpr (BT) bf = frag_bt_f16(B + (size_t)n0 * ldb + k0, ldb);
    else              bf = frag_b_f16(B + (size_t)k0 * ldb + n0, ldb);
    c = wmma_f16(a, bf, c);
  }
  const int nn = n0 + (lane & 15);
  const int mbase = m0 + ((lane >> 4) << 3);
  if constexpr (O16) {
    _Float16* Cp = (_Float16*)C;
#pragma unroll
    for (int v = 0; v < 8; ++v) Cp[(size_t)(mbase + v) * ldc + nn] = (_Float16)c[v];
  } else {
    float* Cp = (float*)C;
#pragma unroll
    for (int v = 0; v < 8; ++v) Cp[(size_t)(mbase + v) * ldc + nn] = c[v];
  }
}

// Heavy KV: per (b,h): C(1024x128) = ctxn(1024x1024) * kv_w[h]^T; split k/v.
__global__ __launch_bounds__(256)
void heavy_kv_gemm_kernel(const _Float16* __restrict__ ctxn,
                          const _Float16* __restrict__ wkv,
                          _Float16* __restrict__ kpre, _Float16* __restrict__ vbuf) {
  const int wv = threadIdx.x >> 5, lane = threadIdx.x & 31;
  const int n0 = blockIdx.x * 16;                       // 0..112
  const int m0 = (blockIdx.y * 8 + wv) * 16;            // token tile
  const int bh = blockIdx.z;
  const int b = bh >> 3, h = bh & 7;
  const _Float16* A  = ctxn + ((size_t)(b * NQ_C + m0)) * DIMC;
  const _Float16* Bt = wkv + (size_t)h * 128 * DIMC + (size_t)n0 * DIMC;
  v8f c = {};
  for (int k0 = 0; k0 < DIMC; k0 += 32)
    c = wmma_f16(frag_a_f16(A + k0, DIMC), frag_bt_f16(Bt + k0, DIMC), c);
  const int n = lane & 15, mb = (lane >> 4) << 3;
  const int col = n0 + n;
#pragma unroll
  for (int v = 0; v < 8; ++v) {
    const int m = m0 + mb + v;
    if (col < DH_C)
      kpre[((size_t)bh * NQ_C + m) * DH_C + col] = (_Float16)c[v];
    else
      vbuf[((size_t)bh * KT_PAD + 1 + m) * DH_C + (col - DH_C)] = (_Float16)c[v];
  }
}

// ------------------------- light (windowed) attention -----------------------

__global__ __launch_bounds__(128)
void light_attn_kernel(const _Float16* __restrict__ qkv, _Float16* __restrict__ lao) {
  __shared__ float sim[WIN_C][3 * WIN_C];               // 64x192 f32 = 48KB
  const int w = blockIdx.x, h = blockIdx.y, b = blockIdx.z;
  const int wv = threadIdx.x >> 5, lane = threadIdx.x & 31;
  const int m0 = wv * 16;
  const int qt0 = w * WIN_C;
  const int kt0 = (w - 1) * WIN_C;
  const size_t tokbase = (size_t)b * SEQ_C;
  // phase 1: sim = (Q * dh^-0.5) @ K^T, masked
  for (int ct = 0; ct < 12; ++ct) {
    v8f c = {};
#pragma unroll
    for (int ks = 0; ks < 2; ++ks) {
      const int k0 = ks * 32;
      const _Float16* Ap = qkv + (tokbase + qt0 + m0) * QKV_C + h * DH_C + k0;
      v16h a = frag_a_f16(Ap, QKV_C);
      v16h bf;
      {
        const int n = lane & 15;
        const int kb = k0 + ((lane >> 4) << 4);
        int kt = kt0 + ct * 16 + n;
        int ktc = kt < 0 ? 0 : (kt > SEQ_C - 1 ? SEQ_C - 1 : kt);
        const _Float16* Kp = qkv + (tokbase + ktc) * QKV_C + HD_C + h * DH_C;
#pragma unroll
        for (int v = 0; v < 8; ++v) {
          bf[2 * v]     = Kp[kb + 2 * v];
          bf[2 * v + 1] = Kp[kb + 2 * v + 1];
        }
      }
      c = wmma_f16(a, bf, c);
    }
    const int n = lane & 15, mb = (lane >> 4) << 3;
    const int jloc = ct * 16 + n;
    const int kt = kt0 + jloc;
#pragma unroll
    for (int v = 0; v < 8; ++v) {
      const int row = m0 + mb + v;
      const int diff = (qt0 + row) - kt;
      const bool valid = (kt >= 0) && (kt < SEQ_C) &&
                         (diff <= WIN_C) && (diff >= -WIN_C);
      sim[row][jloc] = valid ? c[v] * 0.125f : -1e30f;
    }
  }
  __syncthreads();
  // phase 2: softmax, 2 lanes per row
  {
    const int row = m0 + (lane & 15);
    const int c0 = (lane >> 4) * 96;
    float mx = -3.4e38f;
    for (int j = 0; j < 96; ++j) mx = fmaxf(mx, sim[row][c0 + j]);
    mx = fmaxf(mx, __shfl_xor(mx, 16));
    float sum = 0.f;
    for (int j = 0; j < 96; ++j) sum += __expf(sim[row][c0 + j] - mx);
    sum += __shfl_xor(sum, 16);
    const float inv = 1.0f / sum;
    for (int j = 0; j < 96; ++j)
      sim[row][c0 + j] = __expf(sim[row][c0 + j] - mx) * inv;
  }
  __syncthreads();
  // phase 3: out = P @ V
  for (int f = 0; f < 4; ++f) {
    v8f acc = {};
#pragma unroll
    for (int ks = 0; ks < 6; ++ks) {
      v16h a = frag_a_from_f32(&sim[m0][ks * 32], 3 * WIN_C);
      v16h bf;
      {
        const int n = lane & 15;
        const int kb = (lane >> 4) << 4;
#pragma unroll
        for (int v = 0; v < 8; ++v) {
          int j0 = ks * 32 + kb + 2 * v;
          int k1 = kt0 + j0, k2 = kt0 + j0 + 1;
          k1 = k1 < 0 ? 0 : (k1 > SEQ_C - 1 ? SEQ_C - 1 : k1);
          k2 = k2 < 0 ? 0 : (k2 > SEQ_C - 1 ? SEQ_C - 1 : k2);
          const size_t vo = (size_t)(2 * HD_C) + h * DH_C + f * 16 + n;
          bf[2 * v]     = qkv[(tokbase + k1) * QKV_C + vo];
          bf[2 * v + 1] = qkv[(tokbase + k2) * QKV_C + vo];
        }
      }
      acc = wmma_f16(a, bf, acc);
    }
    const int n = lane & 15, mb = (lane >> 4) << 3;
#pragma unroll
    for (int v = 0; v < 8; ++v) {
      const int tok = qt0 + m0 + mb + v;
      lao[(tokbase + tok) * HD_C + h * DH_C + f * 16 + n] = (_Float16)acc[v];
    }
  }
}

// ------------------------- rope + null kv -----------------------------------

__global__ void rope_q_kernel(const _Float16* __restrict__ qh,
                              const int* __restrict__ idxq,
                              _Float16* __restrict__ qhr) {
  const int e = blockIdx.x * 256 + threadIdx.x;
  if (e >= BATCH_C * NQ_C * HD_C) return;
  const int c = e & (HD_C - 1);
  const int bi = e >> 9;
  const int dd = c & 63;
  const int t = idxq[bi];
  const int j = dd & 31;
  const float inv = __powf(10000.0f, -(float)(2 * j) / 64.0f);
  float sp, cp;
  __sincosf((float)t * inv, &sp, &cp);
  const float q = (float)qh[e];
  const float other = (dd < 32) ? -(float)qh[e + 32] : (float)qh[e - 32];
  qhr[e] = (_Float16)(q * cp + other * sp);
}

__global__ void rope_k_kernel(const _Float16* __restrict__ kpre,
                              const int* __restrict__ idxkv,
                              _Float16* __restrict__ kbuf) {
  const int e = blockIdx.x * 256 + threadIdx.x;
  if (e >= BATCH_C * HEADS_C * NQ_C * DH_C) return;
  const int dd = e & 63;
  const int rest = e >> 6;
  const int nt = rest & (NQ_C - 1);
  const int bh = rest >> 10;
  const int b = bh >> 3;
  const int t = idxkv[b * NQ_C + nt];
  const int j = dd & 31;
  const float inv = __powf(10000.0f, -(float)(2 * j) / 64.0f);
  float sp, cp;
  __sincosf((float)t * inv, &sp, &cp);
  const float kv = (float)kpre[e];
  const float other = (dd < 32) ? -(float)kpre[e + 32] : (float)kpre[e - 32];
  kbuf[((size_t)bh * KT_PAD + 1 + nt) * DH_C + dd] = (_Float16)(kv * cp + other * sp);
}

__global__ void null_fill_kernel(const float* __restrict__ null_kv,
                                 _Float16* __restrict__ kbuf,
                                 _Float16* __restrict__ vbuf) {
  const int e = blockIdx.x * 256 + threadIdx.x;
  if (e >= BATCH_C * HEADS_C * DH_C) return;
  const int bh = e >> 6, dd = e & 63;
  const int h = bh & 7;
  kbuf[(size_t)bh * KT_PAD * DH_C + dd] = (_Float16)null_kv[0 * HD_C + h * DH_C + dd];
  vbuf[(size_t)bh * KT_PAD * DH_C + dd] = (_Float16)null_kv[1 * HD_C + h * DH_C + dd];
}

// ------------------------- heavy attention (flash-style) --------------------

__global__ __launch_bounds__(128)
void heavy_attn_kernel(const _Float16* __restrict__ qhr,
                       const _Float16* __restrict__ kbuf,
                       const _Float16* __restrict__ vbuf,
                       _Float16* __restrict__ hao) {
  __shared__ float P[4][16][32];                        // per-wave probs, 8KB
  const int bh = blockIdx.y;
  const int b = bh >> 3, h = bh & 7;
  const int wv = threadIdx.x >> 5, lane = threadIdx.x & 31;
  const int q0 = blockIdx.x * 64 + wv * 16;
  const int n = lane & 15, mb = (lane >> 4) << 3;
  v8f acc[4] = {{}, {}, {}, {}};
  float rm[8], rs[8];
#pragma unroll
  for (int v = 0; v < 8; ++v) { rm[v] = -1e30f; rs[v] = 0.f; }
  for (int p = 0; p < KT_PAD / 32; ++p) {
    const int j0 = p * 32;
    v8f c0 = {}, c1 = {};
#pragma unroll
    for (int ks = 0; ks < 2; ++ks) {
      const int k0 = ks * 32;
      v16h a  = frag_a_f16(qhr + ((size_t)(b * NQ_C + q0)) * HD_C + h * DH_C + k0, HD_C);
      v16h b0 = frag_bt_f16(kbuf + ((size_t)(bh * KT_PAD + j0)) * DH_C + k0, DH_C);
      v16h b1 = frag_bt_f16(kbuf + ((size_t)(bh * KT_PAD + j0 + 16)) * DH_C + k0, DH_C);
      c0 = wmma_f16(a, b0, c0);
      c1 = wmma_f16(a, b1, c1);
    }
    const int jc0 = j0 + n, jc1 = j0 + 16 + n;
#pragma unroll
    for (int v = 0; v < 8; ++v) {
      c0[v] = (jc0 < NQ_C + 1) ? c0[v] * 0.125f : -1e30f;
      c1[v] = (jc1 < NQ_C + 1) ? c1[v] * 0.125f : -1e30f;
    }
#pragma unroll
    for (int v = 0; v < 8; ++v) {
      float mloc = fmaxf(c0[v], c1[v]);
      for (int o = 1; o < 16; o <<= 1) mloc = fmaxf(mloc, __shfl_xor(mloc, o));
      const float mnew = fmaxf(rm[v], mloc);
      const float corr = __expf(rm[v] - mnew);
      const float p0 = __expf(c0[v] - mnew);
      const float p1 = __expf(c1[v] - mnew);
      float sl = p0 + p1;
      for (int o = 1; o < 16; o <<= 1) sl += __shfl_xor(sl, o);
      rs[v] = rs[v] * corr + sl;
      rm[v] = mnew;
#pragma unroll
      for (int f = 0; f < 4; ++f) acc[f][v] *= corr;
      P[wv][mb + v][n]      = p0;
      P[wv][mb + v][16 + n] = p1;
    }
    __syncthreads();
#pragma unroll
    for (int f = 0; f < 4; ++f) {
      v16h a  = frag_a_from_f32(&P[wv][0][0], 32);
      v16h bf = frag_b_f16(vbuf + ((size_t)(bh * KT_PAD + j0)) * DH_C + f * 16, DH_C);
      acc[f] = wmma_f16(a, bf, acc[f]);
    }
    __syncthreads();
  }
#pragma unroll
  for (int v = 0; v < 8; ++v) {
    const float invs = 1.0f / rs[v];
#pragma unroll
    for (int f = 0; f < 4; ++f)
      hao[((size_t)(b * NQ_C + q0 + mb + v)) * HD_C + h * DH_C + f * 16 + n] =
          (_Float16)(acc[f][v] * invs);
  }
}

// ------------------------- scatter / combine --------------------------------

__global__ void map_init_kernel(int* __restrict__ map) {
  const int e = blockIdx.x * 256 + threadIdx.x;
  if (e < BATCH_C * SEQ_C) map[e] = -1;
}

__global__ void map_scatter_kernel(const int* __restrict__ idxq, int* __restrict__ map) {
  const int e = blockIdx.x * 256 + threadIdx.x;
  if (e < BATCH_C * NQ_C) {
    const int b = e >> 10;
    map[b * SEQ_C + idxq[e]] = e & (NQ_C - 1);
  }
}

__global__ __launch_bounds__(256)
void combine_kernel(const float* __restrict__ hfin, const int* __restrict__ map,
                    const float* __restrict__ nullq, float* __restrict__ out) {
  const size_t tok = blockIdx.x;
  const int b = (int)(tok >> 13);
  const int slot = map[tok];
#pragma unroll
  for (int j = 0; j < 4; ++j) {
    const int d = threadIdx.x + j * 256;
    const float add = (slot >= 0)
                          ? hfin[((size_t)(b * NQ_C + slot)) * DIMC + d]
                          : nullq[d];
    out[tok * DIMC + d] += add;
  }
}

// ------------------------- host: pipeline -----------------------------------

extern "C" void kernel_launch(void* const* d_in, const int* in_sizes, int n_in,
                              void* d_out, int out_size, void* d_ws, size_t ws_size,
                              hipStream_t stream) {
  const float* x        = (const float*)d_in[0];
  const float* ln_g     = (const float*)d_in[1];
  const float* ln_b     = (const float*)d_in[2];
  const float* w_qkv    = (const float*)d_in[3];
  const float* w_lo     = (const float*)d_in[4];
  const float* q_tok    = (const float*)d_in[5];
  const float* kv_tok   = (const float*)d_in[6];
  const float* h_gamma  = (const float*)d_in[7];
  const float* w_hq     = (const float*)d_in[8];
  const float* w_hkv    = (const float*)d_in[9];
  const float* null_kv  = (const float*)d_in[10];
  const float* w_ho     = (const float*)d_in[11];
  const float* null_q   = (const float*)d_in[12];
  float* out = (float*)d_out;

  // ---- workspace layout ----
  char* base = (char*)d_ws;
  size_t off = 0;
  auto alloc = [&](size_t bytes) -> char* {
    char* p = base + off;
    off = (off + bytes + 255) & ~(size_t)255;
    return p;
  };
  const size_t TOKS = (size_t)BATCH_C * SEQ_C;          // 16384
  const size_t RT   = (size_t)BATCH_C * NQ_C;           // 2048
  _Float16* xn     = (_Float16*)alloc(TOKS * DIMC * 2);
  _Float16* wqkv16 = (_Float16*)alloc((size_t)DIMC * QKV_C * 2);
  _Float16* wlo16  = (_Float16*)alloc((size_t)HD_C * DIMC * 2);
  _Float16* whq16  = (_Float16*)alloc((size_t)DIMC * HD_C * 2);
  _Float16* wkv16  = (_Float16*)alloc((size_t)HEADS_C * 128 * DIMC * 2);
  _Float16* who16  = (_Float16*)alloc((size_t)HD_C * DIMC * 2);
  _Float16* qkvb   = (_Float16*)alloc(TOKS * QKV_C * 2);
  _Float16* lao    = (_Float16*)alloc(TOKS * HD_C * 2);
  float*    sq     = (float*)alloc(TOKS * 4);
  float*    skv    = (float*)alloc(TOKS * 4);
  float*    scq    = (float*)alloc(TOKS * 4);
  float*    sckv   = (float*)alloc(TOKS * 4);
  int*      idxq   = (int*)alloc(RT * 4);
  int*      idxkv  = (int*)alloc(RT * 4);
  _Float16* rtqn   = (_Float16*)alloc(RT * DIMC * 2);
  _Float16* ctxn   = (_Float16*)alloc(RT * DIMC * 2);
  _Float16* qh     = (_Float16*)alloc(RT * HD_C * 2);
  _Float16* qhr    = (_Float16*)alloc(RT * HD_C * 2);
  _Float16* kpre   = (_Float16*)alloc((size_t)BATCH_C * HEADS_C * NQ_C * DH_C * 2);
  _Float16* kbuf   = (_Float16*)alloc((size_t)BATCH_C * HEADS_C * KT_PAD * DH_C * 2);
  _Float16* vbuf   = (_Float16*)alloc((size_t)BATCH_C * HEADS_C * KT_PAD * DH_C * 2);
  _Float16* hao    = (_Float16*)alloc(RT * HD_C * 2);
  float*    hfin   = (float*)alloc(RT * DIMC * 4);
  int*      map    = (int*)alloc(TOKS * 4);
  (void)ws_size; (void)n_in; (void)in_sizes; (void)out_size;

  auto cvt = [&](const float* s, _Float16* d, int n) {
    f32_to_f16_kernel<<<(n + 255) / 256, 256, 0, stream>>>(s, d, n);
  };
  // 1. weight conversion
  cvt(w_qkv, wqkv16, DIMC * QKV_C);
  cvt(w_lo, wlo16, HD_C * DIMC);
  cvt(w_hq, whq16, DIMC * HD_C);
  cvt(w_hkv, wkv16, HEADS_C * 128 * DIMC);
  cvt(w_ho, who16, HD_C * DIMC);

  // 2. layernorm
  ln_kernel<<<(int)TOKS, 256, 0, stream>>>(x, ln_g, ln_b, xn);

  // 3. light QKV: (16384x1024) @ (1024x1536) -> f16
  gemm_wmma_kernel<false, true><<<dim3(QKV_C / 16, (int)TOKS / 128), 256, 0, stream>>>(
      xn, DIMC, wqkv16, QKV_C, qkvb, QKV_C, DIMC);

  // 4. light local attention
  light_attn_kernel<<<dim3(NW_C, HEADS_C, BATCH_C), 128, 0, stream>>>(qkvb, lao);

  // 5. light out-proj: (16384x512) @ (512x1024) -> f32 d_out
  gemm_wmma_kernel<false, false><<<dim3(DIMC / 16, (int)TOKS / 128), 256, 0, stream>>>(
      lao, HD_C, wlo16, DIMC, out, DIMC, HD_C);

  // 6. routing scores
  route_score_kernel<<<(int)TOKS, 256, 0, stream>>>(x, q_tok, kv_tok, sq, skv);

  // 7. coordinate descent (per batch)
  coor_descent_kernel<<<BATCH_C, 1024, 0, stream>>>(sq, scq);
  coor_descent_kernel<<<BATCH_C, 1024, 0, stream>>>(skv, sckv);

  // 8. top-k selection via bitonic sort
  topk_sort_kernel<<<BATCH_C, 1024, 0, stream>>>(scq, idxq);
  topk_sort_kernel<<<BATCH_C, 1024, 0, stream>>>(sckv, idxkv);

  // 9. gather + rmsnorm routed tokens
  gather_rmsnorm_kernel<<<(int)RT, 256, 0, stream>>>(x, idxq, h_gamma, rtqn);
  gather_rmsnorm_kernel<<<(int)RT, 256, 0, stream>>>(x, idxkv, h_gamma, ctxn);

  // 10. heavy Q: (2048x1024) @ (1024x512) -> f16
  gemm_wmma_kernel<false, true><<<dim3(HD_C / 16, (int)RT / 128), 256, 0, stream>>>(
      rtqn, DIMC, whq16, HD_C, qh, HD_C, DIMC);

  // 11. zero padded K/V buffers
  {
    const int nkv = BATCH_C * HEADS_C * KT_PAD * DH_C;
    zero_f16_kernel<<<(nkv + 255) / 256, 256, 0, stream>>>(kbuf, nkv);
    zero_f16_kernel<<<(nkv + 255) / 256, 256, 0, stream>>>(vbuf, nkv);
  }

  // 12. heavy KV projection (per b,h; split into k-pre-rope and v)
  heavy_kv_gemm_kernel<<<dim3(8, NQ_C / 128, BATCH_C * HEADS_C), 256, 0, stream>>>(
      ctxn, wkv16, kpre, vbuf);

  // 13/14. rope
  rope_q_kernel<<<(BATCH_C * NQ_C * HD_C + 255) / 256, 256, 0, stream>>>(qh, idxq, qhr);
  rope_k_kernel<<<(BATCH_C * HEADS_C * NQ_C * DH_C + 255) / 256, 256, 0, stream>>>(
      kpre, idxkv, kbuf);

  // 15. null kv row 0
  null_fill_kernel<<<(BATCH_C * HEADS_C * DH_C + 255) / 256, 256, 0, stream>>>(
      null_kv, kbuf, vbuf);

  // 16. heavy attention
  heavy_attn_kernel<<<dim3(NQ_C / 64, BATCH_C * HEADS_C), 128, 0, stream>>>(
      qhr, kbuf, vbuf, hao);

  // 17. heavy out-proj: (2048x512) @ (512x1024) -> f32
  gemm_wmma_kernel<false, false><<<dim3(DIMC / 16, (int)RT / 128), 256, 0, stream>>>(
      hao, HD_C, who16, DIMC, hfin, DIMC, HD_C);

  // 18. inverse routing map
  map_init_kernel<<<((int)TOKS + 255) / 256, 256, 0, stream>>>(map);
  map_scatter_kernel<<<((int)RT + 255) / 256, 256, 0, stream>>>(idxq, map);

  // 19. combine: light + (heavy | null token)
  combine_kernel<<<(int)TOKS, 256, 0, stream>>>(hfin, map, null_q, out);
}
